// TSP_Decoder_1649267441901
// MI455X (gfx1250) — matrix operand order repaired and checked by reading, not compiled, not run on hardware
//
#include <hip/hip_runtime.h>
#include <hip/hip_bf16.h>

#define HEADS 8
#define DHEAD 16
#define EMB   128
#define Bx    64
#define Nx    512
#define Px    512

typedef __attribute__((ext_vector_type(16))) __bf16 v16bf;
typedef __attribute__((ext_vector_type(8)))  __bf16 v8bf;
typedef __attribute__((ext_vector_type(8)))  float  v8f;
typedef __attribute__((ext_vector_type(4)))  float  v4f;
typedef __attribute__((ext_vector_type(4)))  unsigned int v4u;
typedef __attribute__((ext_vector_type(8)))  int    v8i;
typedef __attribute__((ext_vector_type(4)))  int    v4i;

#define WMMA_BF16(a, b, c) \
  __builtin_amdgcn_wmma_f32_16x16x32_bf16(false, (a), false, (b), (short)0, (c), false, false)

__device__ __forceinline__ int lane_id() { return (int)(threadIdx.x & 31u); }

__device__ __forceinline__ v16bf concat8(v8bf lo, v8bf hi) {
  return __builtin_shufflevector(lo, hi, 0, 1, 2, 3, 4, 5, 6, 7, 8, 9, 10, 11, 12, 13, 14, 15);
}

// ---- fragment loaders per CDNA5 ISA 7.12.2 (wave32), fully vectorized ----
__device__ __forceinline__ v16bf a_frag_bf(const __bf16* base, int ld) {
  int lane = lane_id();
  const __bf16* r = base + (size_t)(lane & 15) * ld + (lane >> 4) * 8;
  return concat8(*(const v8bf*)r, *(const v8bf*)(r + 16));
}

// A from f32 source with per-row scale folded into the bf16 conversion (softmax 1/sum).
__device__ __forceinline__ v16bf a_frag_f32_scaled(const float* base, int ld,
                                                   const float* rowscale) {
  int lane = lane_id();
  int row = lane & 15;
  float s = rowscale[row];
  const float* r = base + (size_t)row * ld + (lane >> 4) * 8;
  v4f x0 = *(const v4f*)r, x1 = *(const v4f*)(r + 4);
  v4f x2 = *(const v4f*)(r + 16), x3 = *(const v4f*)(r + 20);
  v16bf a;
#pragma unroll
  for (int j = 0; j < 4; ++j) {
    a[j]      = (__bf16)(x0[j] * s);
    a[4 + j]  = (__bf16)(x1[j] * s);
    a[8 + j]  = (__bf16)(x2[j] * s);
    a[12 + j] = (__bf16)(x3[j] * s);
  }
  return a;
}

// B: 32(K)x16(N), N-major source: element (k,n) at base[n*ld+k] -> two 16B runs/lane (k=8h.., 16+8h..).
__device__ __forceinline__ v16bf b_frag(const __bf16* base, int ld) {
  int lane = lane_id();
  const __bf16* c = base + (size_t)(lane & 15) * ld + (lane >> 4) * 8;
  return concat8(*(const v8bf*)c, *(const v8bf*)(c + 16));
}

// B from f32 N-major source (TDM-staged enc tile), cvt to bf16 on load.
__device__ __forceinline__ v16bf b_frag_f32(const float* base, int ld) {
  int lane = lane_id();
  const float* c = base + (size_t)(lane & 15) * ld + (lane >> 4) * 8;
  v4f x0 = *(const v4f*)c, x1 = *(const v4f*)(c + 4);
  v4f x2 = *(const v4f*)(c + 16), x3 = *(const v4f*)(c + 20);
  v16bf b;
#pragma unroll
  for (int j = 0; j < 4; ++j) {
    b[j]      = (__bf16)x0[j];
    b[4 + j]  = (__bf16)x1[j];
    b[8 + j]  = (__bf16)x2[j];
    b[12 + j] = (__bf16)x3[j];
  }
  return b;
}

// C/D 16x16 f32: vgpr r, lane l -> (M = r + 8*(l>=16), N = l&15)
__device__ __forceinline__ void store_d_f32(float* base, int ld, v8f d) {
  int lane = lane_id();
  int n = lane & 15, mh = (lane >> 4) * 8;
#pragma unroll
  for (int r = 0; r < 8; ++r) base[(size_t)(mh + r) * ld + n] = d[r];
}
__device__ __forceinline__ void store_d_bf16(__bf16* base, int ld, v8f d) {
  int lane = lane_id();
  int n = lane & 15, mh = (lane >> 4) * 8;
#pragma unroll
  for (int r = 0; r < 8; ++r) base[(size_t)(mh + r) * ld + n] = (__bf16)d[r];
}

// Stage 128x128 f32 weights into LDS as bf16 TRANSPOSED (coalesced global reads).
__device__ __forceinline__ void stage_wT(__bf16* sWT, const float* W, int tid) {
  for (int i = tid; i < EMB * EMB; i += 128) {
    int k = i >> 7, n = i & 127;
    sWT[(size_t)n * EMB + k] = (__bf16)W[i];
  }
}

// Cooperative LDS(bf16) -> global(bf16) copy in 16-byte chunks (coalesced).
__device__ __forceinline__ void copy_out_bf16(__bf16* dst, const __bf16* src, int n, int tid) {
  for (int c = tid * 8; c < n; c += 128 * 8)
    *(v8bf*)(dst + c) = *(const v8bf*)(src + c);
}

// --- Tensor Data Mover: 2D tile load Global->LDS (ISA 08_async_tensor §8) ---
// dsize_code: 0=1B,1=2B,2=4B. Optional LDS padding via flags (pad_enable/interval/amount).
__device__ __forceinline__ void tdm_load_2d(unsigned lds_off, const void* gptr,
                                            unsigned dsize_code,
                                            unsigned tile_w, unsigned tile_h,
                                            unsigned row_stride,
                                            unsigned tens_w, unsigned tens_h,
                                            unsigned g1_flags) {
  unsigned long long ga = (unsigned long long)gptr;
  v4u g0;
  g0[0] = 1u;                                                 // count=1, user D#
  g0[1] = lds_off;                                            // lds_addr [63:32]
  g0[2] = (unsigned)(ga & 0xffffffffu);                       // global_addr[31:0]
  g0[3] = (unsigned)((ga >> 32) & 0x1ffffffu) | (2u << 30);   // addr[56:32] | type=2
  v8i g1 = {};
  g1[0] = (int)((dsize_code << 16) | g1_flags);               // data_size + pad flags
  g1[1] = (int)((tens_w & 0xffffu) << 16);                    // tensor_dim0[15:0]
  g1[2] = (int)((tens_w >> 16) & 0xffffu) | (int)((tens_h & 0xffffu) << 16);
  g1[3] = (int)((tens_h >> 16) & 0xffffu) | (int)((tile_w & 0xffffu) << 16); // tile_dim0
  g1[4] = (int)(tile_h & 0xffffu);                            // tile_dim1 (tile_dim2=0)
  g1[5] = (int)row_stride;                                    // tensor_dim0_stride[31:0]
  v4i g2 = {};   // 2D: dims 2..4 zero/unused
  v4i g3 = {};
#if __clang_major__ >= 23
  v8i g4 = {};
  __builtin_amdgcn_tensor_load_to_lds(g0, g1, g2, g3, g4, 0);
#else
  __builtin_amdgcn_tensor_load_to_lds(g0, g1, g2, g3, 0);
#endif
}

// pad flags: insert 8 DWORDs (32B) of skipped LDS after every 8 stored DWORDs (32B):
// pad_enable(bit20) | pad_interval=code2->8dw (bits24:22) | pad_amount=code7->8dw (bits31:25)
#define TDM_PAD_32B_EVERY_32B ((1u << 20) | (2u << 22) | (7u << 25))

__device__ __forceinline__ unsigned lds_off_of(const void* p) {
  return (unsigned)(unsigned long long)p;  // LDS aperture: low 32 bits = LDS byte offset
}

// ---------------- kernel 1: K/V projections ----------------
__global__ __launch_bounds__(128, 1) void proj_kv_kernel(
    const float* __restrict__ enc, const float* __restrict__ Wk, const float* __restrict__ Wv,
    __bf16* __restrict__ Kw, __bf16* __restrict__ Vw) {
  extern __shared__ char smem[];
  __bf16* sWkT = (__bf16*)smem;             // 128x128 transposed
  __bf16* sWvT = sWkT + EMB * EMB;          // 128x128 transposed
  __bf16* sA   = sWvT + EMB * EMB;          // 64x128 (A tile, reused as output staging)
  int b = blockIdx.x, r0 = blockIdx.y * 64, tid = (int)threadIdx.x;
  stage_wT(sWkT, Wk, tid);
  stage_wT(sWvT, Wv, tid);
  const float* A = enc + ((size_t)b * Nx + r0) * EMB;
  for (int i = tid; i < 64 * EMB; i += 128) sA[i] = (__bf16)A[i];
  __syncthreads();
  int m0 = (tid >> 5) * 16;
  v16bf af[4];
#pragma unroll
  for (int k = 0; k < 4; ++k) af[k] = a_frag_bf(sA + (size_t)m0 * EMB + k * 32, EMB);
  __syncthreads();  // everyone done reading sA; it becomes output staging
  for (int which = 0; which < 2; ++which) {
    const __bf16* WT = which ? sWvT : sWkT;
    __bf16* Out = which ? Vw : Kw;
    for (int n0 = 0; n0 < EMB; n0 += 16) {
      v8f acc = {};
#pragma unroll
      for (int k = 0; k < 4; ++k)
        acc = WMMA_BF16(af[k], b_frag(WT + (size_t)n0 * EMB + k * 32, EMB), acc);
      store_d_bf16(sA + (size_t)m0 * EMB + n0, EMB, acc);
    }
    __syncthreads();
    copy_out_bf16(Out + ((size_t)b * Nx + r0) * EMB, sA, 64 * EMB, tid);
    __syncthreads();
  }
}

// ---------------- kernel 2: fused Q projection ----------------
__global__ __launch_bounds__(128, 1) void proj_q_kernel(
    const float* __restrict__ q1, const float* __restrict__ last,
    const float* __restrict__ Wqf, const float* __restrict__ Wql, __bf16* __restrict__ Qw) {
  extern __shared__ char smem[];
  __bf16* sWfT = (__bf16*)smem;
  __bf16* sWlT = sWfT + EMB * EMB;
  __bf16* sA1  = sWlT + EMB * EMB;   // 64x128 (reused as staging)
  __bf16* sA2  = sA1 + 64 * EMB;     // 64x128
  int b = blockIdx.x, r0 = blockIdx.y * 64, tid = (int)threadIdx.x;
  stage_wT(sWfT, Wqf, tid);
  stage_wT(sWlT, Wql, tid);
  const float* A1 = q1 + ((size_t)b * Px + r0) * EMB;
  const float* A2 = last + ((size_t)b * Px + r0) * EMB;
  for (int i = tid; i < 64 * EMB; i += 128) { sA1[i] = (__bf16)A1[i]; sA2[i] = (__bf16)A2[i]; }
  __syncthreads();
  int m0 = (tid >> 5) * 16;
  v16bf af1[4], af2[4];
#pragma unroll
  for (int k = 0; k < 4; ++k) {
    af1[k] = a_frag_bf(sA1 + (size_t)m0 * EMB + k * 32, EMB);
    af2[k] = a_frag_bf(sA2 + (size_t)m0 * EMB + k * 32, EMB);
  }
  __syncthreads();
  for (int n0 = 0; n0 < EMB; n0 += 16) {
    v8f acc = {};
#pragma unroll
    for (int k = 0; k < 4; ++k) {
      acc = WMMA_BF16(af1[k], b_frag(sWfT + (size_t)n0 * EMB + k * 32, EMB), acc);
      acc = WMMA_BF16(af2[k], b_frag(sWlT + (size_t)n0 * EMB + k * 32, EMB), acc);
    }
    store_d_bf16(sA1 + (size_t)m0 * EMB + n0, EMB, acc);
  }
  __syncthreads();
  copy_out_bf16(Qw + ((size_t)b * Px + r0) * EMB, sA1, 64 * EMB, tid);
}

// ---------------- kernel 3: multi-head attention (scores in LDS, TDM staging) ----------------
__global__ __launch_bounds__(128, 1) void attn_kernel(
    const __bf16* __restrict__ Qw, const __bf16* __restrict__ Kw,
    const __bf16* __restrict__ Vw, const float* __restrict__ mask, __bf16* __restrict__ AO) {
  extern __shared__ char smem[];
  float*  S    = (float*)smem;                               // 64 x 512 scores/probs
  __bf16* sQ   = (__bf16*)(smem + (size_t)64 * Nx * 4);      // 64 x 32 (TDM-padded)
  __bf16* sK   = sQ + 64 * 32;                               // 64 x 32 (TDM-padded)
  __bf16* sVt  = sK + 64 * 32;                               // 16 x 512 (V^T [d][node])
  float*  sInv = (float*)(sVt + DHEAD * Nx);                 // 64 row 1/sum
  int b = blockIdx.x, h = blockIdx.y, p0 = blockIdx.z * 64;
  int tid = (int)threadIdx.x;
  // pre-zero the padded Q/K tiles once; TDM pad skips the zero gaps thereafter
  for (int i = tid; i < 2 * 64 * 32; i += 128) sQ[i] = (__bf16)0.0f;
  __syncthreads();
  if (tid < 32) {  // wave 0 drives the Tensor Data Mover (EXEC ignored by TDM)
    tdm_load_2d(lds_off_of(sQ), Qw + ((size_t)b * Px + p0) * EMB + h * DHEAD,
                /*dsize=*/1, /*tile_w=*/DHEAD, /*tile_h=*/64,
                /*row_stride=*/EMB, /*tens_w=*/DHEAD, /*tens_h=*/Px, TDM_PAD_32B_EVERY_32B);
    __builtin_amdgcn_s_wait_tensorcnt(0);
  }
  __syncthreads();
  int m0 = (tid >> 5) * 16;
  v16bf aq = a_frag_bf(sQ + (size_t)m0 * 32, 32);            // invariant over N chunks
  for (int nc = 0; nc < Nx; nc += 64) {
    __syncthreads();
    if (tid < 32) {
      tdm_load_2d(lds_off_of(sK), Kw + ((size_t)b * Nx + nc) * EMB + h * DHEAD,
                  /*dsize=*/1, /*tile_w=*/DHEAD, /*tile_h=*/64,
                  /*row_stride=*/EMB, /*tens_w=*/DHEAD, /*tens_h=*/Nx, TDM_PAD_32B_EVERY_32B);
      __builtin_amdgcn_s_wait_tensorcnt(0);
    }
    __syncthreads();
#pragma unroll
    for (int t = 0; t < 4; ++t) {
      v8f acc = {};
      acc = WMMA_BF16(aq, b_frag(sK + (size_t)(t * 16) * 32, 32), acc);
      store_d_f32(S + (size_t)m0 * Nx + nc + t * 16, Nx, acc);
    }
  }
  // stage V transposed: sVt[d][node], coalesced global reads over d
  for (int i = tid; i < DHEAD * Nx; i += 128) {
    int node = i >> 4, d = i & 15;
    sVt[(size_t)d * Nx + node] = Vw[((size_t)b * Nx + node) * EMB + h * DHEAD + d];
  }
  __syncthreads();
  // phase A: coalesced scale + mask into S
  {
    const float* mbase = mask + ((size_t)b * Px + p0) * Nx;
    for (int i = tid; i < 64 * Nx; i += 128) S[i] = S[i] * 0.25f + mbase[i];
  }
  __syncthreads();
  // phase B: per-row max/exp/sum (LDS-only); 1/sum folded into AV A-fragments
  if (tid < 64) {
    float* row = S + (size_t)tid * Nx;
    float mx = -3.4e38f;
    for (int n = 0; n < Nx; ++n) mx = fmaxf(mx, row[n]);
    float sum = 0.0f;
    for (int n = 0; n < Nx; ++n) { float e = __expf(row[n] - mx); row[n] = e; sum += e; }
    sInv[tid] = 1.0f / sum;
  }
  __syncthreads();
  v8f acc = {};
  for (int nc = 0; nc < Nx; nc += 32)  // AV: K-dim = 32 nodes per WMMA, fully dense
    acc = WMMA_BF16(a_frag_f32_scaled(S + (size_t)m0 * Nx + nc, Nx, sInv + m0),
                    b_frag(sVt + nc, Nx), acc);
  store_d_bf16(sQ + (size_t)m0 * DHEAD, DHEAD, acc);  // stage 64x16 tile in LDS
  __syncthreads();
  {  // coalesced 16B stores of the head slice (free concat into [B,P,128])
    int r = tid >> 1, half = tid & 1;
    *(v8bf*)(AO + ((size_t)b * Px + p0 + r) * EMB + h * DHEAD + half * 8) =
        *(const v8bf*)(sQ + (size_t)r * DHEAD + half * 8);
  }
}

// ---------------- kernel 4: output projection + gate MLP ----------------
__global__ __launch_bounds__(128, 1) void mh_gate_kernel(
    const __bf16* __restrict__ AO, const float* __restrict__ Wc, const float* __restrict__ bc,
    const float* __restrict__ g1w, const float* __restrict__ g1b,
    const float* __restrict__ g2w, const float* __restrict__ g2b,
    __bf16* __restrict__ MH, float* __restrict__ LAM) {
  extern __shared__ char smem[];
  __bf16* sWT = (__bf16*)smem;                              // 128x128 transposed
  float*  sMH = (float*)(smem + (size_t)EMB * EMB * 2);     // 64x128
  float*  sG1 = sMH + 64 * EMB;                             // 128x64
  int b = blockIdx.x, p0 = blockIdx.y * 64, tid = (int)threadIdx.x;
  stage_wT(sWT, Wc, tid);
  for (int i = tid; i < EMB * (EMB / 2); i += 128) sG1[i] = g1w[i];
  __syncthreads();
  int m0 = (tid >> 5) * 16;
  v16bf af[4];
#pragma unroll
  for (int k = 0; k < 4; ++k)
    af[k] = a_frag_bf(AO + ((size_t)b * Px + p0 + m0) * EMB + k * 32, EMB);
  for (int n0 = 0; n0 < EMB; n0 += 16) {
    v8f acc = {};
#pragma unroll
    for (int k = 0; k < 4; ++k)
      acc = WMMA_BF16(af[k], b_frag(sWT + (size_t)n0 * EMB + k * 32, EMB), acc);
    int lane = tid & 31, n = lane & 15, mh = (lane >> 4) * 8;
    float bias = bc[n0 + n];
#pragma unroll
    for (int r = 0; r < 8; ++r)
      sMH[(size_t)(m0 + mh + r) * EMB + n0 + n] = acc[r] + bias;
  }
  __syncthreads();
  // coalesced bf16 write-out of mh (16B chunks)
  {
    __bf16* dst = MH + ((size_t)b * Px + p0) * EMB;
    for (int c = tid * 8; c < 64 * EMB; c += 128 * 8) {
      v4f x0 = *(const v4f*)(sMH + c), x1 = *(const v4f*)(sMH + c + 4);
      v8bf o;
#pragma unroll
      for (int j = 0; j < 4; ++j) { o[j] = (__bf16)x0[j]; o[4 + j] = (__bf16)x1[j]; }
      *(v8bf*)(dst + c) = o;
    }
  }
  if (tid < 64) {  // lam = sigmoid(relu(mh@g1w+g1b)@g2w+g2b): tiny, LDS-resident
    const float* row = sMH + (size_t)tid * EMB;
    float lacc = g2b[0];
    for (int j = 0; j < EMB / 2; ++j) {
      float hsum = g1b[j];
      for (int e = 0; e < EMB; ++e) hsum += row[e] * sG1[(size_t)e * (EMB / 2) + j];
      lacc += fmaxf(hsum, 0.0f) * g2w[j];
    }
    LAM[(size_t)b * Px + p0 + tid] = 1.0f / (1.0f + __expf(-lacc));
  }
}

// ---------------- kernel 5: single-head score + clip + bias + softmax ----------------
__global__ __launch_bounds__(128, 1) void final_kernel(
    const __bf16* __restrict__ MH, const float* __restrict__ enc,
    const float* __restrict__ conf, const float* __restrict__ mask,
    const float* __restrict__ LAM, float* __restrict__ out) {
  extern __shared__ char smem[];
  float* S    = (float*)smem;                                // 64 x 512
  float* sEf  = (float*)(smem + (size_t)64 * Nx * 4);        // 64 x 128 enc chunk (f32, TDM)
  float* sLam = sEf + 64 * EMB;                              // 64
  float* sInv = sLam + 64;                                   // 64
  int b = blockIdx.x, p0 = blockIdx.y * 64, tid = (int)threadIdx.x;
  if (tid < 64) sLam[tid] = LAM[(size_t)b * Px + p0 + tid];
  int m0 = (tid >> 5) * 16;
  v16bf afr[4];
#pragma unroll
  for (int k = 0; k < 4; ++k)
    afr[k] = a_frag_bf(MH + ((size_t)b * Px + p0 + m0) * EMB + k * 32, EMB);
  unsigned sEf_lds = lds_off_of(sEf);
  for (int nc = 0; nc < Nx; nc += 64) {
    __syncthreads();
    if (tid < 32) {  // wave 0 drives the Tensor Data Mover
      tdm_load_2d(sEf_lds, enc + ((size_t)b * Nx + nc) * EMB,
                  /*dsize=*/2, /*tile_w=*/EMB, /*tile_h=*/64,
                  /*row_stride=*/EMB, /*tens_w=*/EMB, /*tens_h=*/Nx, 0u);
      __builtin_amdgcn_s_wait_tensorcnt(0);
    }
    __syncthreads();
#pragma unroll
    for (int t = 0; t < 4; ++t) {
      v8f acc = {};
#pragma unroll
      for (int k = 0; k < 4; ++k)
        acc = WMMA_BF16(afr[k], b_frag_f32(sEf + (size_t)(t * 16) * EMB + k * 32, EMB), acc);
      store_d_f32(S + (size_t)m0 * Nx + nc + t * 16, Nx, acc);
    }
  }
  __syncthreads();
  // phase A: coalesced tanh-clip + gated cluster bias + mask
  {
    const float* mbase = mask + ((size_t)b * Px + p0) * Nx;
    const float* crow  = conf + (size_t)b * Nx;
    for (int i = tid; i < 64 * Nx; i += 128) {
      int row = i >> 9, col = i & (Nx - 1);
      S[i] = 10.0f * tanhf(S[i] * 0.08838834764831845f) + sLam[row] * crow[col] + mbase[i];
    }
  }
  __syncthreads();
  // phase B: per-row max/exp/sum (LDS-only)
  if (tid < 64) {
    float* row = S + (size_t)tid * Nx;
    float mx = -3.4e38f;
    for (int n = 0; n < Nx; ++n) mx = fmaxf(mx, row[n]);
    float sum = 0.0f;
    for (int n = 0; n < Nx; ++n) { float e = __expf(row[n] - mx); row[n] = e; sum += e; }
    sInv[tid] = 1.0f / sum;
  }
  __syncthreads();
  // phase C: coalesced normalize + store
  {
    float* obase = out + ((size_t)b * Px + p0) * Nx;
    for (int i = tid; i < 64 * Nx; i += 128) obase[i] = S[i] * sInv[i >> 9];
  }
}

extern "C" void kernel_launch(void* const* d_in, const int* in_sizes, int n_in,
                              void* d_out, int out_size, void* d_ws, size_t ws_size,
                              hipStream_t stream) {
  (void)in_sizes; (void)n_in; (void)out_size; (void)ws_size;
  const float* enc  = (const float*)d_in[0];
  const float* q1   = (const float*)d_in[1];
  const float* last = (const float*)d_in[2];
  const float* mask = (const float*)d_in[3];
  const float* conf = (const float*)d_in[4];
  const float* Wqf  = (const float*)d_in[5];
  const float* Wql  = (const float*)d_in[6];
  const float* Wk   = (const float*)d_in[7];
  const float* Wv   = (const float*)d_in[8];
  const float* Wc   = (const float*)d_in[9];
  const float* bc   = (const float*)d_in[10];
  const float* g1w  = (const float*)d_in[11];
  const float* g1b  = (const float*)d_in[12];
  const float* g2w  = (const float*)d_in[13];
  const float* g2b  = (const float*)d_in[14];
  float* out = (float*)d_out;

  char* ws = (char*)d_ws;
  size_t off = 0;
  __bf16* Kw  = (__bf16*)(ws + off); off += (size_t)Bx * Nx * EMB * 2;
  __bf16* Vw  = (__bf16*)(ws + off); off += (size_t)Bx * Nx * EMB * 2;
  __bf16* Qw  = (__bf16*)(ws + off); off += (size_t)Bx * Px * EMB * 2;
  __bf16* AO  = (__bf16*)(ws + off); off += (size_t)Bx * Px * EMB * 2;
  __bf16* MH  = (__bf16*)(ws + off); off += (size_t)Bx * Px * EMB * 2;
  float*  LAM = (float*)(ws + off);  off += (size_t)Bx * Px * 4;

  dim3 blk(128);  // 4 waves (wave32)
  proj_kv_kernel<<<dim3(Bx, Nx / 64), blk, 2 * EMB * EMB * 2 + 64 * EMB * 2, stream>>>(
      enc, Wk, Wv, Kw, Vw);
  proj_q_kernel<<<dim3(Bx, Px / 64), blk, 2 * EMB * EMB * 2 + 2 * 64 * EMB * 2, stream>>>(
      q1, last, Wqf, Wql, Qw);
  attn_kernel<<<dim3(Bx, HEADS, Px / 64), blk,
                64 * Nx * 4 + 2 * 64 * 32 * 2 + DHEAD * Nx * 2 + 64 * 4, stream>>>(
      Qw, Kw, Vw, mask, AO);
  mh_gate_kernel<<<dim3(Bx, Px / 64), blk,
                   EMB * EMB * 2 + 64 * EMB * 4 + EMB * (EMB / 2) * 4, stream>>>(
      AO, Wc, bc, g1w, g1b, g2w, g2b, MH, LAM);
  final_kernel<<<dim3(Bx, Px / 64), blk,
                 64 * Nx * 4 + 64 * EMB * 4 + 2 * 64 * 4, stream>>>(
      MH, enc, conf, mask, LAM, out);
}
